// MetaR_86801289052573
// MI455X (gfx1250) — matrix operand order repaired and checked by reading, not compiled, not run on hardware
//
#include <hip/hip_runtime.h>
#include <hip/hip_bf16.h>

typedef _Float16 v16h __attribute__((ext_vector_type(16)));
typedef _Float16 v8h  __attribute__((ext_vector_type(8)));
typedef float    v8f  __attribute__((ext_vector_type(8)));

#define BATCH 4096
#define MM    200          // neighbors
#define DD    100          // embed dim
#define NENTC 5000
#define K2    200          // 2*D (concat width)
#define MP    208          // 13*16 padded M
#define DP    112          // 7*16  padded D
#define KP    224          // 7*32  padded K

__global__ __launch_bounds__(256)
void metar_fused_kernel(const int*   __restrict__ conn,    // (B, M, 3)
                        const int*   __restrict__ target,  // (B, 1, 2)
                        const float* __restrict__ symb,    // (NSYM, D)
                        const float* __restrict__ coent,   // (NENT, NENT)
                        const float* __restrict__ gcnW,    // (D, 2D)
                        const float* __restrict__ gcnWb,   // (D,)
                        const float* __restrict__ gcnB,    // (D,)
                        const float* __restrict__ attnW,   // (1, D)
                        const float* __restrict__ attnWb,  // (1,)
                        const float* __restrict__ gateW,   // (1, D)
                        const float* __restrict__ gateWb,  // (1,)
                        const float* __restrict__ gateB,   // (1,)
                        float*       __restrict__ out)     // (B, D)
{
    __shared__ _Float16 sA[MP][KP];     // concat [rel | ent], f16, zero padded
    __shared__ _Float16 sW[DP][KP];     // gcn weights, f16, zero padded
    __shared__ float    sOut[MP][DP];   // post-leaky-relu activations
    __shared__ float    sBias[DP];
    __shared__ float    sAttnWv[DP];
    __shared__ float    sOutAttn[DP];
    __shared__ float    sNeigh[DP];
    __shared__ float    sLogit[MP];     // attn logits -> softmax weights
    __shared__ float    sCoW[MP];
    __shared__ float    sRed[256];
    __shared__ float    sScal[4];

    const int b    = blockIdx.x;
    const int tid  = threadIdx.x;
    const int lane = tid & 31;
    const int wave = tid >> 5;
    const int hi   = lane >> 4;   // lane half (0/1)
    const int ln   = lane & 15;

    const int entself   = conn[(b * MM + 0) * 3 + 0];
    const int targetEnt = target[b * 2 + 0];

    // ---------------- Phase 1: stage operands into LDS (f32 -> f16) ----------
    for (int i = tid; i < MP * KP; i += 256) {
        const int m = i / KP, k = i % KP;
        float v = 0.0f;
        if (m < MM && k < K2) {
            const int idx = (k < DD) ? conn[(b * MM + m) * 3 + 1]
                                     : conn[(b * MM + m) * 3 + 2];
            const int kk  = (k < DD) ? k : (k - DD);
            v = symb[idx * DD + kk];
        }
        sA[m][k] = (_Float16)v;
    }
    for (int i = tid; i < DP * KP; i += 256) {
        const int d = i / KP, k = i % KP;
        const float v = (d < DD && k < K2) ? gcnW[d * K2 + k] : 0.0f;
        sW[d][k] = (_Float16)v;
    }
    for (int i = tid; i < DP; i += 256) {
        sBias[i]   = (i < DD) ? (gcnWb[i] + gcnB[i]) : 0.0f;
        sAttnWv[i] = (i < DD) ? attnW[i] : 0.0f;
    }
    for (int m = tid; m < MP; m += 256) {
        sCoW[m] = (m < MM)
                ? coent[conn[(b * MM + m) * 3 + 2] * NENTC + targetEnt]
                : 0.0f;
    }
    __syncthreads();

    // ---------------- Phase 2: GEMM via v_wmma_f32_16x16x32_f16 --------------
    // out[m,d] = leakyrelu( sum_k concat[m,k] * W[d,k] + bias[d] )
    for (int tile = wave; tile < 13 * 7; tile += 8) {
        const int m0 = (tile % 13) * 16;
        const int d0 = (tile / 13) * 16;
        v8f acc = {};
        #pragma unroll
        for (int kk = 0; kk < 7; ++kk) {
            const int kb = kk * 32;
            union { v16h v; v8h h[2]; } a, bm;
            // A (16x32 f16): lane half 0 -> K {0..7,16..23}; half 1 -> {8..15,24..31}
            a.h[0]  = *(const v8h*)&sA[m0 + ln][kb + 8 * hi];
            a.h[1]  = *(const v8h*)&sA[m0 + ln][kb + 16 + 8 * hi];
            // B (32x16 f16): col = lane&15; lane half 0 -> K {0..7,16..23}; half 1 -> {8..15,24..31}
            bm.h[0] = *(const v8h*)&sW[d0 + ln][kb + 8 * hi];
            bm.h[1] = *(const v8h*)&sW[d0 + ln][kb + 16 + 8 * hi];
            acc = __builtin_amdgcn_wmma_f32_16x16x32_f16(
                false, a.v, false, bm.v, (short)0, acc, false, false);
        }
        // C/D layout: VGPR v -> row m0 + v + 8*hi, col d0 + ln
        const float bias = sBias[d0 + ln];
        #pragma unroll
        for (int v = 0; v < 8; ++v) {
            float x = acc[v] + bias;
            x = (x > 0.0f) ? x : 0.01f * x;
            sOut[m0 + v + 8 * hi][d0 + ln] = x;
        }
    }
    __syncthreads();

    // ---------------- Phase 3: attention logits ------------------------------
    for (int m = tid; m < MM; m += 256) {
        float s = attnWb[0];
        for (int d = 0; d < DD; ++d) s += sOut[m][d] * sAttnWv[d];
        sLogit[m] = s;
    }
    __syncthreads();

    // ---------------- Phase 4: softmax over m --------------------------------
    const float lv = (tid < MM) ? sLogit[tid] : -1e30f;
    sRed[tid] = lv;
    __syncthreads();
    for (int s = 128; s > 0; s >>= 1) {
        if (tid < s) sRed[tid] = fmaxf(sRed[tid], sRed[tid + s]);
        __syncthreads();
    }
    const float mx = sRed[0];
    __syncthreads();
    const float ev = (tid < MM) ? expf(lv - mx) : 0.0f;
    sRed[tid] = ev;
    __syncthreads();
    for (int s = 128; s > 0; s >>= 1) {
        if (tid < s) sRed[tid] += sRed[tid + s];
        __syncthreads();
    }
    const float inv = 1.0f / sRed[0];
    __syncthreads();
    if (tid < MM) sLogit[tid] = ev * inv;   // normalized attention weights
    __syncthreads();

    // ---------------- Phase 5: attn-weighted sum + neighbor embedding --------
    for (int d = tid; d < DD; d += 256) {
        float oa = 0.0f, ne = 0.0f;
        for (int m = 0; m < MM; ++m) {
            oa += sOut[m][d] * sLogit[m];
            ne += sCoW[m] * (float)sA[m][DD + d];   // ent half of concat
        }
        sOutAttn[d] = oa;
        sNeigh[d]   = ne;
    }
    __syncthreads();

    // ---------------- Phase 6: gate ------------------------------------------
    if (tid == 0) {
        float g = gateWb[0] + gateB[0];
        for (int d = 0; d < DD; ++d) g += sOutAttn[d] * gateW[d];
        sScal[0] = 1.0f / (1.0f + expf(-g));
    }
    __syncthreads();
    const float gate = sScal[0];

    // ---------------- Phase 7: combine and write -----------------------------
    for (int d = tid; d < DD; d += 256) {
        const float self = symb[entself * DD + d];
        out[b * DD + d] = sOutAttn[d] * gate + self * (1.0f - gate) + sNeigh[d];
    }
}

extern "C" void kernel_launch(void* const* d_in, const int* in_sizes, int n_in,
                              void* d_out, int out_size, void* d_ws, size_t ws_size,
                              hipStream_t stream) {
    const int*   conn   = (const int*)  d_in[0];
    const int*   target = (const int*)  d_in[1];
    const float* symb   = (const float*)d_in[2];
    const float* coent  = (const float*)d_in[3];
    const float* gcnW   = (const float*)d_in[4];
    const float* gcnWb  = (const float*)d_in[5];
    const float* gcnB   = (const float*)d_in[6];
    const float* attnW  = (const float*)d_in[7];
    const float* attnWb = (const float*)d_in[8];
    const float* gateW  = (const float*)d_in[9];
    const float* gateWb = (const float*)d_in[10];
    const float* gateB  = (const float*)d_in[11];

    metar_fused_kernel<<<BATCH, 256, 0, stream>>>(
        conn, target, symb, coent, gcnW, gcnWb, gcnB,
        attnW, attnWb, gateW, gateWb, gateB, (float*)d_out);
}
